// CausalSelfAttention_77996606095756
// MI455X (gfx1250) — compile-verified, matrix operand order checked by reading, not run on hardware
//
#include <hip/hip_runtime.h>

#define B_  4
#define T_  2048
#define C_  1024
#define H_  16
#define D_  64
#define C3_ 3072
#define BT_ (B_ * T_)

typedef __attribute__((ext_vector_type(16))) __bf16 bf16x16;
typedef __attribute__((ext_vector_type(8)))  __bf16 bf16x8;
typedef __attribute__((ext_vector_type(8)))  float  f32x8;

// GCC-style generic vector to match the builtin's 'V4i' pointee exactly.
typedef int async_v4i __attribute__((vector_size(16)));

// ---- CDNA5 async global->LDS staging (guarded; falls back to sync copy) ----
#if defined(__has_builtin)
#  if __has_builtin(__builtin_amdgcn_global_load_async_to_lds_b128)
#    define HAVE_ASYNC_LDS 1
#  endif
#  if __has_builtin(__builtin_amdgcn_s_wait_asynccnt)
#    define HAVE_WAIT_ASYNC 1
#  endif
#endif
#ifndef HAVE_ASYNC_LDS
#  define HAVE_ASYNC_LDS 0
#endif
#ifndef HAVE_WAIT_ASYNC
#  define HAVE_WAIT_ASYNC 0
#endif

#if HAVE_WAIT_ASYNC && HAVE_ASYNC_LDS
#  define WAIT_ASYNC0() __builtin_amdgcn_s_wait_asynccnt(0)
#else
#  define WAIT_ASYNC0() ((void)0)
#endif

// Copy 16 bytes global -> LDS (async GLOBAL_LOAD_ASYNC_TO_LDS_B128 on CDNA5).
static __device__ __forceinline__ void stage16(const __bf16* g, __bf16* l) {
#if HAVE_ASYNC_LDS
    // Builtin signature (from hipcc diagnostic): (v4i32 AS1*, v4i32 AS3*, Imm i32, Imm i32)
    __builtin_amdgcn_global_load_async_to_lds_b128(
        (__attribute__((address_space(1))) async_v4i*)const_cast<__bf16*>(g),
        (__attribute__((address_space(3))) async_v4i*)l,
        0, 0);
#else
    *reinterpret_cast<bf16x8*>(l) = *reinterpret_cast<const bf16x8*>(g);
#endif
}

// Load one wave-striped 16-bit A/B fragment (16 bf16 per lane) for
// v_wmma_f32_16x16x32_bf16. Caller passes row_base + k0 + (lane>>4)*8.
// ISA layout: VGPR0..3 hold K = half*8..+7, VGPR4..7 hold K = 16+half*8..+7.
static __device__ __forceinline__ bf16x16 load_frag(const __bf16* p) {
    bf16x8 lo = *reinterpret_cast<const bf16x8*>(p);
    bf16x8 hi = *reinterpret_cast<const bf16x8*>(p + 16);
    return __builtin_shufflevector(lo, hi, 0,1,2,3,4,5,6,7,8,9,10,11,12,13,14,15);
}

static __device__ __forceinline__ f32x8 wmma_bf16(bf16x16 a, bf16x16 b, f32x8 c) {
    return __builtin_amdgcn_wmma_f32_16x16x32_bf16(false, a, false, b, (short)0, c, false, false);
}

// ---------------- prep kernels ----------------

__global__ void cvt_f32_bf16(const float* __restrict__ in, __bf16* __restrict__ out, int n) {
    int i = blockIdx.x * blockDim.x + threadIdx.x;
    if (i < n) out[i] = (__bf16)in[i];
}

// W [K][N] f32 -> Wt [N][K] bf16
__global__ void transpose_cvt(const float* __restrict__ W, __bf16* __restrict__ Wt, int K, int N) {
    int i = blockIdx.x * blockDim.x + threadIdx.x;
    if (i >= K * N) return;
    int n = i / K, k = i % K;
    Wt[(size_t)n * K + k] = (__bf16)W[(size_t)k * N + n];
}

// qkv bf16 [BT][3C] -> RoPE'd q,k [bh][t][d] and transposed v [bh][d][t]
__global__ void rope_scatter(const __bf16* __restrict__ qkv,
                             __bf16* __restrict__ qr, __bf16* __restrict__ kr,
                             __bf16* __restrict__ vT) {
    int idx = blockIdx.x * blockDim.x + threadIdx.x;   // (b,t,h,i) i = rotation pair 0..31
    if (idx >= B_ * T_ * H_ * 32) return;
    int i = idx & 31;
    int tmp = idx >> 5;
    int h = tmp & 15; tmp >>= 4;
    int t = tmp & (T_ - 1);
    int b = tmp >> 11;
    int bh = b * H_ + h;
    const size_t row = ((size_t)b * T_ + t) * (size_t)C3_;
    int col = h * D_ + 2 * i;
    float inv = __expf((-2.0f * (float)i / (float)D_) * 9.210340371976184f); // 10000^(-2i/D)
    float ang = (float)t * inv;
    float sn = __sinf(ang), cs = __cosf(ang);

    size_t qo = ((size_t)bh * T_ + t) * D_ + 2 * i;
    float x1 = (float)qkv[row + col], x2 = (float)qkv[row + col + 1];
    qr[qo]     = (__bf16)(x1 * cs - x2 * sn);
    qr[qo + 1] = (__bf16)(x1 * sn + x2 * cs);

    x1 = (float)qkv[row + C_ + col]; x2 = (float)qkv[row + C_ + col + 1];
    kr[qo]     = (__bf16)(x1 * cs - x2 * sn);
    kr[qo + 1] = (__bf16)(x1 * sn + x2 * cs);

    vT[((size_t)bh * D_ + 2 * i)     * T_ + t] = qkv[row + 2 * C_ + col];
    vT[((size_t)bh * D_ + 2 * i + 1) * T_ + t] = qkv[row + 2 * C_ + col + 1];
}

// ---------------- WMMA GEMM with LDS-staged 128x128 block tiles ----------------
// C[M][N] = A[M][K] * Bt[N][K]^T + bias.
// 256 threads = 8 waves; each wave computes 32(M) x 64(N); K-loop step 32 with
// double-buffered LDS tiles staged via async global->LDS copies.
#define MBLK  128
#define NBLK  128
#define KSTEP 32
#define LDSP  40   // padded LDS row stride (elements) = 80B, kills bank conflicts

__global__ __launch_bounds__(256)
void gemm_bf16_wmma(const __bf16* __restrict__ A, const __bf16* __restrict__ Bt,
                    const float* __restrict__ bias,
                    float* __restrict__ outF, __bf16* __restrict__ outB,
                    int M, int N, int K) {
    __shared__ __align__(16) __bf16 As[2][MBLK][LDSP];   // 20 KB
    __shared__ __align__(16) __bf16 Bs[2][NBLK][LDSP];   // 20 KB

    const int tid  = threadIdx.x;
    const int lane = tid & 31, wave = tid >> 5;
    const int half = lane >> 4, l16 = lane & 15;

    const int tilesX = N / NBLK;
    const int gm0 = (blockIdx.x / tilesX) * MBLK;
    const int gn0 = (blockIdx.x % tilesX) * NBLK;

    const int wm = (wave & 3) * 32;   // wave M offset: 0,32,64,96
    const int wn = (wave >> 2) * 64;  // wave N offset: 0,64

    f32x8 acc[2][4] = {};

    // staging assignment: 128 rows x 4 chunks(8 elems) = 512 copies, 2 per thread per matrix
    const int srow   = tid >> 2;         // 0..63
    const int schunk = (tid & 3) * 8;    // 0,8,16,24

    auto stageTile = [&](int kb, int buf) {
        const int k0 = kb * KSTEP;
        const __bf16* ga = A + (size_t)(gm0 + srow) * K + k0 + schunk;
        stage16(ga,                  &As[buf][srow][schunk]);
        stage16(ga + (size_t)64 * K, &As[buf][64 + srow][schunk]);
        const __bf16* gb = Bt + (size_t)(gn0 + srow) * K + k0 + schunk;
        stage16(gb,                  &Bs[buf][srow][schunk]);
        stage16(gb + (size_t)64 * K, &Bs[buf][64 + srow][schunk]);
    };

    const int nk = K / KSTEP;
    stageTile(0, 0);
    WAIT_ASYNC0();
    __syncthreads();

    for (int kb = 0; kb < nk; ++kb) {
        const int cur = kb & 1;
        if (kb + 1 < nk) stageTile(kb + 1, cur ^ 1);   // prefetch next tile

        bf16x16 af[2], bfr[4];
        #pragma unroll
        for (int i = 0; i < 2; ++i)
            af[i] = load_frag(&As[cur][wm + i * 16 + l16][half * 8]);
        #pragma unroll
        for (int j = 0; j < 4; ++j)
            bfr[j] = load_frag(&Bs[cur][wn + j * 16 + l16][half * 8]);

        #pragma unroll
        for (int i = 0; i < 2; ++i)
            #pragma unroll
            for (int j = 0; j < 4; ++j)
                acc[i][j] = wmma_bf16(af[i], bfr[j], acc[i][j]);

        WAIT_ASYNC0();     // next tile landed in LDS
        __syncthreads();   // everyone done reading cur before it is overwritten
    }

    // C/D layout: lanes 0-15 -> M = r, N = lane; lanes 16-31 -> M = 8+r.
    #pragma unroll
    for (int i = 0; i < 2; ++i)
    #pragma unroll
    for (int j = 0; j < 4; ++j) {
        int n = gn0 + wn + j * 16 + l16;
        float bv = bias ? bias[n] : 0.0f;
        #pragma unroll
        for (int r = 0; r < 8; ++r) {
            int m = gm0 + wm + i * 16 + r + 8 * half;
            float v = acc[i][j][r] + bv;
            if (outF) outF[(size_t)m * N + n] = v;
            else      outB[(size_t)m * N + n] = (__bf16)v;
        }
    }
}

// ---------------- flash attention (WMMA, online softmax) ----------------
// Grid: (B*H) * (T/64) blocks of 128 threads; each wave owns one 16-query tile.
__global__ __launch_bounds__(128)
void flash_attn_wmma(const __bf16* __restrict__ qr, const __bf16* __restrict__ kr,
                     const __bf16* __restrict__ vT, __bf16* __restrict__ yb) {
    __shared__ __align__(16) __bf16 plds[4][16][32];   // per-wave P tile staging

    const int lane = threadIdx.x & 31;
    const int wave = threadIdx.x >> 5;
    const int half = lane >> 4, l16 = lane & 15;

    const int bh = blockIdx.x >> 5;                    // T/64 = 32 q-chunks per (b,h)
    const int q0 = ((blockIdx.x & 31) * 4 + wave) * 16;
    const int b = bh >> 4, h = bh & 15;

    const __bf16* Q  = qr + (size_t)bh * T_ * D_;
    const __bf16* Kk = kr + (size_t)bh * T_ * D_;
    const __bf16* V  = vT + (size_t)bh * D_ * T_;      // [d][t]

    // Q fragments stay resident across the whole key loop.
    bf16x16 qf0, qf1;
    {
        const __bf16* qrow = Q + (size_t)(q0 + l16) * D_ + half * 8;
        qf0 = load_frag(qrow);        // d = 0..31
        qf1 = load_frag(qrow + 32);   // d = 32..63
    }

    f32x8 acc[4] = {};                // y tile 16 x 64 (4 n-subtiles)
    float rmax[8], rsum[8];
    #pragma unroll
    for (int r = 0; r < 8; ++r) { rmax[r] = -3.0e38f; rsum[r] = 0.0f; }
    const float scale = 0.125f;       // 1/sqrt(64)

    const int kblocks = (q0 + 47) >> 5;    // keys 0 .. q0+15, blocks of 32
    for (int kb = 0; kb < kblocks; ++kb) {
        const int k0 = kb * 32;

        // S = Q K^T : two 16x16 n-halves, D=64 -> 2 WMMA K-steps each
        f32x8 s[2] = {};
        #pragma unroll
        for (int nh = 0; nh < 2; ++nh) {
            const __bf16* krow = Kk + (size_t)(k0 + nh * 16 + l16) * D_ + half * 8;
            s[nh] = wmma_bf16(qf0, load_frag(krow),      s[nh]);
            s[nh] = wmma_bf16(qf1, load_frag(krow + 32), s[nh]);
        }

        // scale + causal mask + online softmax; stage P into LDS (layout swap).
        #pragma unroll
        for (int r = 0; r < 8; ++r) {
            int m = q0 + r + 8 * half;             // global query row
            float v0 = s[0][r] * scale;
            float v1 = s[1][r] * scale;
            if (k0 + l16      > m) v0 = -3.0e38f;
            if (k0 + 16 + l16 > m) v1 = -3.0e38f;

            float mx = fmaxf(v0, v1);
            #pragma unroll
            for (int off = 1; off < 16; off <<= 1) mx = fmaxf(mx, __shfl_xor(mx, off, 16));
            float mnew = fmaxf(rmax[r], mx);
            float a    = __expf(rmax[r] - mnew);
            rmax[r] = mnew;

            float p0 = __expf(v0 - mnew);
            float p1 = __expf(v1 - mnew);
            float ps = p0 + p1;
            #pragma unroll
            for (int off = 1; off < 16; off <<= 1) ps += __shfl_xor(ps, off, 16);
            rsum[r] = rsum[r] * a + ps;

            plds[wave][r + 8 * half][l16]      = (__bf16)p0;
            plds[wave][r + 8 * half][16 + l16] = (__bf16)p1;
            #pragma unroll
            for (int j = 0; j < 4; ++j) acc[j][r] *= a;
        }
        __builtin_amdgcn_wave_barrier();   // keep DS store -> DS load ordering

        // Re-read P as a 16x32 A-fragment.
        bf16x16 pf;
        {
            const __bf16* prow = &plds[wave][l16][half * 8];
            bf16x8 lo = *reinterpret_cast<const bf16x8*>(prow);
            bf16x8 hi = *reinterpret_cast<const bf16x8*>(prow + 16);
            pf = __builtin_shufflevector(lo, hi, 0,1,2,3,4,5,6,7,8,9,10,11,12,13,14,15);
        }
        __builtin_amdgcn_wave_barrier();

        // acc += P (16x32) * V (32x64); vT rows are contiguous over keys.
        #pragma unroll
        for (int j = 0; j < 4; ++j) {
            const __bf16* vrow = V + (size_t)(j * 16 + l16) * T_ + k0 + half * 8;
            acc[j] = wmma_bf16(pf, load_frag(vrow), acc[j]);
        }
    }

    // Normalize and scatter: y[b][t][h*64+d], bf16 for the proj GEMM.
    #pragma unroll
    for (int j = 0; j < 4; ++j) {
        #pragma unroll
        for (int r = 0; r < 8; ++r) {
            int t = q0 + r + 8 * half;
            int d = j * 16 + l16;
            float v = acc[j][r] / rsum[r];
            yb[((size_t)b * T_ + t) * C_ + h * D_ + d] = (__bf16)v;
        }
    }
}

// ---------------- launch ----------------

extern "C" void kernel_launch(void* const* d_in, const int* in_sizes, int n_in,
                              void* d_out, int out_size, void* d_ws, size_t ws_size,
                              hipStream_t stream) {
    const float* x      = (const float*)d_in[0];
    const float* W_qkv  = (const float*)d_in[1];
    const float* b_qkv  = (const float*)d_in[2];
    const float* W_proj = (const float*)d_in[3];
    const float* b_proj = (const float*)d_in[4];
    float* out = (float*)d_out;

    char* ws = (char*)d_ws;
    size_t off = 0;
    auto alloc = [&](size_t bytes) -> void* {
        void* p = ws + off;
        off += (bytes + 255) & ~(size_t)255;
        return p;
    };

    __bf16* xb     = (__bf16*)alloc((size_t)BT_ * C_  * 2);   // x in bf16 (reused as y later)
    __bf16* wtqkv  = (__bf16*)alloc((size_t)C3_ * C_  * 2);   // W_qkv^T bf16
    __bf16* wtproj = (__bf16*)alloc((size_t)C_  * C_  * 2);   // W_proj^T bf16
    __bf16* qkv    = (__bf16*)alloc((size_t)BT_ * C3_ * 2);   // qkv bf16
    __bf16* qrp    = (__bf16*)alloc((size_t)BT_ * C_  * 2);   // q rope [bh][t][d]
    __bf16* krp    = (__bf16*)alloc((size_t)BT_ * C_  * 2);   // k rope [bh][t][d]
    __bf16* vTp    = (__bf16*)alloc((size_t)BT_ * C_  * 2);   // v^T   [bh][d][t]
    __bf16* yb     = xb;                                      // alias: x-bf16 retired after GEMM1

    // 1) convert x -> bf16
    {
        int n = BT_ * C_;
        cvt_f32_bf16<<<(n + 255) / 256, 256, 0, stream>>>(x, xb, n);
    }
    // 2) transpose+convert weights
    {
        int n = C_ * C3_;
        transpose_cvt<<<(n + 255) / 256, 256, 0, stream>>>(W_qkv, wtqkv, C_, C3_);
        n = C_ * C_;
        transpose_cvt<<<(n + 255) / 256, 256, 0, stream>>>(W_proj, wtproj, C_, C_);
    }
    // 3) GEMM1: qkv = x @ W_qkv + b_qkv   (M=8192, N=3072, K=1024), bf16 out
    {
        int blocks = (BT_ / MBLK) * (C3_ / NBLK);   // 64 * 24
        gemm_bf16_wmma<<<blocks, 256, 0, stream>>>(xb, wtqkv, b_qkv,
                                                   (float*)nullptr, qkv,
                                                   BT_, C3_, C_);
    }
    // 4) RoPE + head scatter (+ v transpose)
    {
        int n = B_ * T_ * H_ * 32;
        rope_scatter<<<(n + 255) / 256, 256, 0, stream>>>(qkv, qrp, krp, vTp);
    }
    // 5) flash attention -> y bf16 [B][T][C]
    {
        flash_attn_wmma<<<B_ * H_ * (T_ / 64), 128, 0, stream>>>(qrp, krp, vTp, yb);
    }
    // 6) GEMM2: out = y @ W_proj + b_proj  (M=8192, N=1024, K=1024), f32 out
    {
        int blocks = (BT_ / MBLK) * (C_ / NBLK);    // 64 * 8
        gemm_bf16_wmma<<<blocks, 256, 0, stream>>>(yb, wtproj, b_proj,
                                                   out, (__bf16*)nullptr,
                                                   BT_, C_, C_);
    }
}
